// VQMetaBaseline_76665166234108
// MI455X (gfx1250) — compile-verified, matrix operand order checked by reading, not run on hardware
//
#include <hip/hip_runtime.h>
#include <hip/hip_bf16.h>

typedef __attribute__((ext_vector_type(16))) __bf16 v16bf;
typedef __attribute__((ext_vector_type(8)))  float  v8f;

__device__ __forceinline__ float lrelu_f(float v) { return v >= 0.f ? v : 0.1f * v; }

// float -> bf16, round to nearest even (explicit, no reliance on implicit cvt)
__device__ __forceinline__ __bf16 f2bf(float f) {
  unsigned u = __builtin_bit_cast(unsigned, f);
  unsigned r = (u + 0x7FFFu + ((u >> 16) & 1u)) >> 16;
  unsigned short us = (unsigned short)r;
  return __builtin_bit_cast(__bf16, us);
}

// ---------------------------------------------------------------------------
// Implicit-GEMM conv / conv-transpose / FC kernel using CDNA5 WMMA (bf16).
// Block = 256 threads (8 waves) computing a 16(cout) x 512(pixel) tile:
//   - A (weights, 16x32 bf16) staged once per block, shared by all 8 waves
//   - each thread owns TWO output-pixel columns of B; pixel decode is pure
//     shift/mask (all spatial dims are powers of two) and hoisted out of K
//   - each wave holds 4 independent accumulators -> 4 back-to-back
//     v_wmma_f32_16x16x32_bf16 issues per K-step (XDL pipelining)
// Template params make kk2d / dilation / upsample compile-time so the
// divides strength-reduce and the branches fold:
//   KSZ  : kernel size (1,3,4)
//   DIL  : lhs_dilation for conv-transpose (1 or 2)
//   UPIN : nearest-x2 upsample fused on the input (1 or 2)
//   TMODE: weight stored [Cin,Cout,k,k], flipped (ConvTranspose2d layout)
// Runtime flags:
//   post : 0 none | 1 bn->act | 2 act->bn | 3 act only
//   act  : 0 none | 1 leaky-relu(0.1) | 2 relu
// ---------------------------------------------------------------------------
template<int KSZ, int DIL, int UPIN, int TMODE>
__global__ __launch_bounds__(256) void conv_wmma_kernel(
    const float* __restrict__ x, const float* __restrict__ w,
    const float* __restrict__ bias,
    const float* __restrict__ bng, const float* __restrict__ bnb,
    const float* __restrict__ bnm, const float* __restrict__ bnv,
    float* __restrict__ out,
    int Nn, int Cin, int Hin, int Win, int pad,
    int Cout, int Ho, int Wo, int woSh, int howoSh, int post, int act)
{
  constexpr int KK2D = KSZ * KSZ;
  const int tid  = threadIdx.x;
  const int lane = tid & 31;
  const int wv   = tid >> 5;            // wave id 0..7
  const int Hup = Hin * UPIN, Wup = Win * UPIN;
  const int Ktot = Cin * KK2D;
  const int HoWo = 1 << howoSh;
  const int NP = Nn << howoSh;
  const int m0 = blockIdx.y * 16;
  const int pblk = blockIdx.x * 512;

  __shared__ __bf16 As[16 * 32];        // weights tile  [m][k]
  __shared__ __bf16 Bs[32][32 * 16];    // 32 pixel tiles of [k][n]

  // two owned B columns per thread; decode hoisted (shift/mask only)
  int cimg[2], coy[2], cox[2]; bool cval[2];
#pragma unroll
  for (int c = 0; c < 2; ++c) {
    int p = pblk + tid + c * 256;
    cval[c] = p < NP;
    cimg[c] = p >> howoSh;
    int rem = p & (HoWo - 1);
    coy[c] = rem >> woSh;
    cox[c] = rem & (Wo - 1);
  }
  const int tcol_tile = tid >> 4;       // 0..15 (second column: +16)
  const int tcol_n    = tid & 15;

  v8f acc[4];
#pragma unroll
  for (int q = 0; q < 4; ++q) acc[q] = (v8f){0.f,0.f,0.f,0.f,0.f,0.f,0.f,0.f};

  for (int k0 = 0; k0 < Ktot; k0 += 32) {
    // prefetch next weight chunk (global_prefetch_b8)
    __builtin_prefetch(w + (size_t)m0 * Ktot + k0 + 32, 0, 3);

    // ---- A staging: 512 elements over 256 threads (2 each) ----
#pragma unroll
    for (int t = 0; t < 2; ++t) {
      int idx = tid * 2 + t;
      int mi = idx >> 5, kk = idx & 31;
      int kg = k0 + kk;
      int m = m0 + mi;
      float av = 0.f;
      if (m < Cout && kg < Ktot) {
        int ci = kg / KK2D, rs = kg % KK2D;
        int r = rs / KSZ, s = rs % KSZ;
        if (TMODE)
          av = w[(((size_t)ci * Cout + m) * KK2D) + (size_t)(KSZ - 1 - r) * KSZ + (KSZ - 1 - s)];
        else
          av = w[(size_t)m * Ktot + kg];
      }
      As[mi * 32 + kk] = f2bf(av);
    }

    // ---- B staging: two 32-deep columns per thread (im2col gather) ----
#pragma unroll
    for (int c = 0; c < 2; ++c) {
      for (int kb = 0; kb < 32; ++kb) {
        int kg = k0 + kb;
        float bv = 0.f;
        if (cval[c] && kg < Ktot) {
          int ci = kg / KK2D, rs = kg % KK2D;
          int r = rs / KSZ, s = rs % KSZ;
          int iyd = coy[c] - pad + r, ixd = cox[c] - pad + s;
          bool okk = true;
          if (DIL == 2) {               // lhs-dilation (conv-transpose)
            okk = (iyd >= 0) && (ixd >= 0) && !(iyd & 1) && !(ixd & 1);
            iyd >>= 1; ixd >>= 1;
          }
          if (okk && iyd >= 0 && ixd >= 0 && iyd < Hup && ixd < Wup) {
            int iy = (UPIN == 2) ? (iyd >> 1) : iyd;   // fused nearest up2
            int ix = (UPIN == 2) ? (ixd >> 1) : ixd;
            bv = x[(((size_t)cimg[c] * Cin + ci) * Hin + iy) * Win + ix];
          }
        }
        Bs[tcol_tile + c * 16][kb * 16 + tcol_n] = f2bf(bv);
      }
    }
    __syncthreads();

    // ---- fragments + 4 independent WMMAs per wave ----
    int half = lane >> 4, mr = lane & 15;
    v16bf a, b[4];
#pragma unroll
    for (int j = 0; j < 8; ++j) {
      a[j]     = As[mr * 32 + half * 8 + j];
      a[j + 8] = As[mr * 32 + 16 + half * 8 + j];
#pragma unroll
      for (int q = 0; q < 4; ++q) {
        b[q][j]     = Bs[wv * 4 + q][(half * 8 + j) * 16 + mr];
        b[q][j + 8] = Bs[wv * 4 + q][(16 + half * 8 + j) * 16 + mr];
      }
    }
#pragma unroll
    for (int q = 0; q < 4; ++q)
      acc[q] = __builtin_amdgcn_wmma_f32_16x16x32_bf16(false, a, false, b[q],
                                                       (short)0, acc[q], false, false);
    __syncthreads();
  }

  // ---- epilogue: C/D layout -> VGPR r holds M = r + 8*(lane/16), N = lane%16
  int half = lane >> 4, nn = lane & 15;
#pragma unroll
  for (int q = 0; q < 4; ++q) {
    int p = pblk + (wv * 4 + q) * 16 + nn;
    if (p >= NP) continue;
    int img = p >> howoSh, rem = p & (HoWo - 1);
    int oy = rem >> woSh, ox = rem & (Wo - 1);
#pragma unroll
    for (int r = 0; r < 8; ++r) {
      int m = m0 + half * 8 + r;
      if (m >= Cout) continue;
      float v = acc[q][r];
      if (bias) v += bias[m];
      if (post == 1) {          // bn -> act
        float sc = bng[m] * rsqrtf(bnv[m] + 1e-5f);
        v = (v - bnm[m]) * sc + bnb[m];
        if (act == 1) v = lrelu_f(v); else if (act == 2) v = fmaxf(v, 0.f);
      } else if (post == 2) {   // act -> bn
        if (act == 1) v = lrelu_f(v); else if (act == 2) v = fmaxf(v, 0.f);
        float sc = bng[m] * rsqrtf(bnv[m] + 1e-5f);
        v = (v - bnm[m]) * sc + bnb[m];
      } else if (post == 3) {   // act only
        if (act == 1) v = lrelu_f(v); else if (act == 2) v = fmaxf(v, 0.f);
      }
      out[(((size_t)img * Cout + m) * Ho + oy) * Wo + ox] = v;
    }
  }
}

// out = maxpool2( lrelu(a + b) )
__global__ void respool_kernel(const float* __restrict__ a, const float* __restrict__ b,
                               float* __restrict__ out, int N, int C, int Hin, int Win)
{
  int Ho = Hin >> 1, Wo = Win >> 1;
  int tot = N * C * Ho * Wo;
  int i = blockIdx.x * blockDim.x + threadIdx.x;
  if (i >= tot) return;
  int ox = i % Wo; int t = i / Wo;
  int oy = t % Ho; t /= Ho;
  int c = t % C;   int n = t / C;
  size_t base = (((size_t)n * C + c) * Hin + oy * 2) * Win + ox * 2;
  float m = -3.4e38f;
#pragma unroll
  for (int dy = 0; dy < 2; ++dy)
#pragma unroll
    for (int dx = 0; dx < 2; ++dx) {
      size_t o = base + (size_t)dy * Win + dx;
      float v = lrelu_f(a[o] + b[o]);
      m = fmaxf(m, v);
    }
  out[i] = m;
}

// [N,C,H,W] -> [N,C] spatial mean
__global__ void meanpool_kernel(const float* __restrict__ in, float* __restrict__ out,
                                int NC, int HW)
{
  int i = blockIdx.x * blockDim.x + threadIdx.x;
  if (i >= NC) return;
  const float* p = in + (size_t)i * HW;
  float s = 0.f;
  for (int j = 0; j < HW; ++j) s += p[j];
  out[i] = s / (float)HW;
}

// per-row argmin over codebook (K=4096, D=128), gather q, partial ||q-z||^2
__global__ __launch_bounds__(256) void vq_assign_kernel(
    const float* __restrict__ z, const float* __restrict__ emb,
    float* __restrict__ q, int* __restrict__ idx, float* __restrict__ lossp)
{
  const int n = blockIdx.x, t = threadIdx.x;
  __shared__ float sd[256];
  __shared__ int   si[256];
  float best = 3.4e38f; int bi = 0;
  for (int k = t; k < 4096; k += 256) {
    float d = 0.f;
    const float* e = emb + (size_t)k * 128;
    const float* zr = z + (size_t)n * 128;
    for (int j = 0; j < 128; ++j) { float df = zr[j] - e[j]; d += df * df; }
    if (d < best) { best = d; bi = k; }
  }
  sd[t] = best; si[t] = bi;
  __syncthreads();
  for (int s = 128; s > 0; s >>= 1) {
    if (t < s) {
      if (sd[t + s] < sd[t] || (sd[t + s] == sd[t] && si[t + s] < si[t])) {
        sd[t] = sd[t + s]; si[t] = si[t + s];
      }
    }
    __syncthreads();
  }
  int wsel = si[0];
  if (t == 0) idx[n] = wsel;
  if (t < 128) q[(size_t)n * 128 + t] = emb[(size_t)wsel * 128 + t];
  __syncthreads();
  float dv = 0.f;
  if (t < 128) { float df = emb[(size_t)wsel * 128 + t] - z[(size_t)n * 128 + t]; dv = df * df; }
  sd[t] = dv;
  __syncthreads();
  for (int s = 128; s > 0; s >>= 1) {
    if (t < s) sd[t] += sd[t + s];
    __syncthreads();
  }
  if (t == 0) lossp[n] = sd[0];
}

// outLP[0]=commit loss, outLP[1]=perplexity
__global__ void vq_finalize_kernel(const int* __restrict__ idx,
                                   const float* __restrict__ lossp,
                                   float* __restrict__ outLP)
{
  float s = 0.f;
  for (int i = 0; i < 8; ++i) s += lossp[i];
  outLP[0] = 0.25f * s / 1024.f;   // mean over 8*128, * commit
  float ent = 0.f;
  for (int i = 0; i < 8; ++i) {
    bool first = true;
    for (int j = 0; j < i; ++j) if (idx[j] == idx[i]) { first = false; break; }
    if (first) {
      int c = 0;
      for (int j = 0; j < 8; ++j) if (idx[j] == idx[i]) ++c;
      float pr = (float)c / 8.f;
      ent += pr * logf(pr + 1e-10f);
    }
  }
  outLP[1] = expf(-ent);
}

// prototypes (way=2, shot=2) + cosine logits * temp for 4 queries
__global__ void proto_kernel(const float* __restrict__ q, const float* __restrict__ temp,
                             float* __restrict__ logits)
{
  int t = threadIdx.x;
  if (t >= 8) return;
  int qi = t >> 1, w = t & 1;
  float dot = 0.f, np = 0.f, nq = 0.f;
  for (int d = 0; d < 128; ++d) {
    float pv = 0.5f * (q[(size_t)(w * 2) * 128 + d] + q[(size_t)(w * 2 + 1) * 128 + d]);
    float qv = q[(size_t)(4 + qi) * 128 + d];
    dot += pv * qv; np += pv * pv; nq += qv * qv;
  }
  float den = fmaxf(sqrtf(np), 1e-12f) * fmaxf(sqrtf(nq), 1e-12f);
  logits[qi * 2 + w] = dot / den * temp[0];
}

// ---------------------------------------------------------------------------
static inline int ilog2i(int v) { return 31 - __builtin_clz((unsigned)v); }

static void launch_conv(hipStream_t st, const float* x, const float* w, const float* bias,
                        const float* g, const float* b, const float* m, const float* v,
                        float* out, int Nn, int Cin, int Hin, int Win, int upin, int dil,
                        int pad, int ksz, int Cout, int Ho, int Wo, int tmode, int post, int act)
{
  int NP = Nn * Ho * Wo;
  int woSh = ilog2i(Wo), howoSh = ilog2i(Ho * Wo);
  dim3 grid((NP + 511) / 512, (Cout + 15) / 16);
  dim3 blk(256);
  if (ksz == 3 && upin == 1) {
    hipLaunchKernelGGL((conv_wmma_kernel<3,1,1,0>), grid, blk, 0, st,
                       x, w, bias, g, b, m, v, out,
                       Nn, Cin, Hin, Win, pad, Cout, Ho, Wo, woSh, howoSh, post, act);
  } else if (ksz == 1) {
    hipLaunchKernelGGL((conv_wmma_kernel<1,1,1,0>), grid, blk, 0, st,
                       x, w, bias, g, b, m, v, out,
                       Nn, Cin, Hin, Win, pad, Cout, Ho, Wo, woSh, howoSh, post, act);
  } else if (ksz == 4) {  // conv-transpose, fused up2, dil 2, flipped weights
    hipLaunchKernelGGL((conv_wmma_kernel<4,2,2,1>), grid, blk, 0, st,
                       x, w, bias, g, b, m, v, out,
                       Nn, Cin, Hin, Win, pad, Cout, Ho, Wo, woSh, howoSh, post, act);
  } else {                // decoder 3x3 conv with fused up2
    hipLaunchKernelGGL((conv_wmma_kernel<3,1,2,0>), grid, blk, 0, st,
                       x, w, bias, g, b, m, v, out,
                       Nn, Cin, Hin, Win, pad, Cout, Ho, Wo, woSh, howoSh, post, act);
  }
  (void)dil;
}

extern "C" void kernel_launch(void* const* d_in, const int* in_sizes, int n_in,
                              void* d_out, int out_size, void* d_ws, size_t ws_size,
                              hipStream_t stream)
{
  (void)in_sizes; (void)n_in; (void)out_size; (void)ws_size;
#define IN(i) ((const float*)d_in[(i)])
  // Input index map (depth-first insertion order of setup_inputs pytree):
  // 0:x_shot 1:x_query
  // enc blocks i=0..4 at base=2+i*20: w1,bn1(g,b,m,v),w2,bn2x4,w3,bn3x4,wd,bndx4
  // 102:fc_w 103:fc_b
  // 104:dfc_w 105:dfc_b 106-109:bn 110:w7 111:b7 112:w6 113:b6 114-117:bn6
  // 118:w5 119-122:bn5 123:w4 124-127:bn4 128:w3 129-132:bn3 133:w2 134-137:bn2
  // 138:w1 139:codebook 140:temp
  float* ws = (float*)d_ws;
  float* B1 = ws;
  float* B2 = B1 + 67108864ULL;      // 8*32*512*512
  float* B3 = B2 + 67108864ULL;
  float* P  = B3 + 67108864ULL;      // pooled [8,512]
  float* Z  = P + 4096;              // [8,128]
  float* Q  = Z + 1024;              // [8,128]
  float* LP = Q + 1024;              // [8] partial losses
  int*  IDX = (int*)(LP + 16);
  float* D0 = (float*)(IDX + 16);    // dec fc out [8,512]
  float* OUT = (float*)d_out;
  // out layout: logits[8] | recon[2097152] | z[1024] | q[1024] | loss | perp
  const size_t OFF_RECON = 8, OFF_Z = 8 + 2097152ULL, OFF_Q = OFF_Z + 1024, OFF_LP = OFF_Q + 1024;

  // concat x_shot(4 imgs) + x_query(4 imgs) -> B1 [8,1,512,512]
  hipMemcpyAsync(B1, d_in[0], (size_t)4 * 512 * 512 * sizeof(float),
                 hipMemcpyDeviceToDevice, stream);
  hipMemcpyAsync(B1 + (size_t)4 * 512 * 512, d_in[1], (size_t)4 * 512 * 512 * sizeof(float),
                 hipMemcpyDeviceToDevice, stream);

  // -------- encoder: 5 ResNet stages --------
  int chans[6] = {1, 32, 64, 128, 256, 512};
  int Hs = 512;
  float *cur = B1, *t1 = B2, *t2 = B3;
  for (int blk = 0; blk < 5; ++blk) {
    int cin = chans[blk], cout = chans[blk + 1], base = 2 + blk * 20;
    // lrelu(bn(conv1(x)))
    launch_conv(stream, cur, IN(base + 0), nullptr,
                IN(base + 1), IN(base + 2), IN(base + 3), IN(base + 4),
                t1, 8, cin, Hs, Hs, 1, 1, 1, 3, cout, Hs, Hs, 0, 1, 1);
    // lrelu(bn(conv2(.)))
    launch_conv(stream, t1, IN(base + 5), nullptr,
                IN(base + 6), IN(base + 7), IN(base + 8), IN(base + 9),
                t2, 8, cout, Hs, Hs, 1, 1, 1, 3, cout, Hs, Hs, 0, 1, 1);
    // bn(conv3(.))
    launch_conv(stream, t2, IN(base + 10), nullptr,
                IN(base + 11), IN(base + 12), IN(base + 13), IN(base + 14),
                t1, 8, cout, Hs, Hs, 1, 1, 1, 3, cout, Hs, Hs, 0, 1, 0);
    // bn(conv1x1(x)) shortcut
    launch_conv(stream, cur, IN(base + 15), nullptr,
                IN(base + 16), IN(base + 17), IN(base + 18), IN(base + 19),
                t2, 8, cin, Hs, Hs, 1, 1, 0, 1, cout, Hs, Hs, 0, 1, 0);
    // maxpool2(lrelu(out + idn)) -> cur
    int Ho = Hs / 2, tot = 8 * cout * Ho * Ho;
    hipLaunchKernelGGL(respool_kernel, dim3((tot + 255) / 256), dim3(256), 0, stream,
                       t1, t2, cur, 8, cout, Hs, Hs);
    Hs = Ho;
  }
  // global mean pool [8,512,16,16] -> P[8,512]
  hipLaunchKernelGGL(meanpool_kernel, dim3(16), dim3(256), 0, stream, cur, P, 8 * 512, Hs * Hs);
  // z = P @ fc_w.T + fc_b   (FC as 1x1 conv GEMM)
  launch_conv(stream, P, IN(102), IN(103), nullptr, nullptr, nullptr, nullptr,
              Z, 8, 512, 1, 1, 1, 1, 0, 1, 128, 1, 1, 0, 0, 0);

  // -------- VQ --------
  hipLaunchKernelGGL(vq_assign_kernel, dim3(8), dim3(256), 0, stream, Z, IN(139), Q, IDX, LP);
  hipLaunchKernelGGL(vq_finalize_kernel, dim3(1), dim3(1), 0, stream, IDX, LP, OUT + OFF_LP);
  hipLaunchKernelGGL(proto_kernel, dim3(1), dim3(8), 0, stream, Q, IN(140), OUT);
  hipMemcpyAsync(OUT + OFF_Z, Z, 1024 * sizeof(float), hipMemcpyDeviceToDevice, stream);
  hipMemcpyAsync(OUT + OFF_Q, Q, 1024 * sizeof(float), hipMemcpyDeviceToDevice, stream);

  // -------- decoder --------
  // relu(bn(q @ dfc_w.T + dfc_b)) -> D0 [8,512] viewed as [8,512,1,1]
  launch_conv(stream, Q, IN(104), IN(105), IN(106), IN(107), IN(108), IN(109),
              D0, 8, 128, 1, 1, 1, 1, 0, 1, 512, 1, 1, 0, 1, 2);
  // up2 -> ct(w7)+b7 -> lrelu : [8,512,4,4]
  launch_conv(stream, D0, IN(110), IN(111), nullptr, nullptr, nullptr, nullptr,
              B1, 8, 512, 1, 1, 2, 2, 2, 4, 512, 4, 4, 1, 3, 1);
  // up2 -> bn6(lrelu(ct(w6)+b6)) : [8,512,16,16]
  launch_conv(stream, B1, IN(112), IN(113), IN(114), IN(115), IN(116), IN(117),
              B2, 8, 512, 4, 4, 2, 2, 2, 4, 512, 16, 16, 1, 2, 1);
  // up2 -> bn5(lrelu(conv w5)) : [8,256,32,32]
  launch_conv(stream, B2, IN(118), nullptr, IN(119), IN(120), IN(121), IN(122),
              B1, 8, 512, 16, 16, 2, 1, 1, 3, 256, 32, 32, 0, 2, 1);
  // up2 -> bn4(lrelu(conv w4)) : [8,128,64,64]
  launch_conv(stream, B1, IN(123), nullptr, IN(124), IN(125), IN(126), IN(127),
              B2, 8, 256, 32, 32, 2, 1, 1, 3, 128, 64, 64, 0, 2, 1);
  // up2 -> bn3(lrelu(conv w3)) : [8,64,128,128]
  launch_conv(stream, B2, IN(128), nullptr, IN(129), IN(130), IN(131), IN(132),
              B1, 8, 128, 64, 64, 2, 1, 1, 3, 64, 128, 128, 0, 2, 1);
  // up2 -> bn2(lrelu(conv w2)) : [8,32,256,256]
  launch_conv(stream, B1, IN(133), nullptr, IN(134), IN(135), IN(136), IN(137),
              B2, 8, 64, 128, 128, 2, 1, 1, 3, 32, 256, 256, 0, 2, 1);
  // up2 -> conv w1 : [8,1,512,512] directly into d_out recon slot
  launch_conv(stream, B2, IN(138), nullptr, nullptr, nullptr, nullptr, nullptr,
              OUT + OFF_RECON, 8, 32, 256, 256, 2, 1, 1, 3, 1, 512, 512, 0, 0, 0);
#undef IN
}
